// TemporalTransformerBCI_51539607552838
// MI455X (gfx1250) — matrix operand order repaired and checked by reading, not compiled, not run on hardware
//
#include <hip/hip_runtime.h>
#include <hip/hip_bf16.h>
#include <math.h>

// ---------------------------------------------------------------------------
// CDNA5 / gfx1250 WMMA types
// ---------------------------------------------------------------------------
typedef __attribute__((ext_vector_type(16))) __bf16 v16bf;
typedef __attribute__((ext_vector_type(2)))  __bf16 v2bf;
typedef __attribute__((ext_vector_type(8)))  float  v8f;

union BF16Frag { v16bf v; unsigned u[8]; };

// fp32 pair -> packed bf16 dword, ONE VALU instruction.
//  - v_cvt_pk_bf16_f32 (RNE) when the builtin exists
//  - else v_perm_b32 (truncation: grab high 16 bits of each fp32)
__device__ inline unsigned pack2bf(float a, float b) {
#if __has_builtin(__builtin_amdgcn_cvt_pk_bf16_f32)
    v2bf p = __builtin_amdgcn_cvt_pk_bf16_f32(a, b);
    return __builtin_bit_cast(unsigned, p);
#else
    return __builtin_amdgcn_perm(__builtin_bit_cast(unsigned, b),
                                 __builtin_bit_cast(unsigned, a), 0x07060302u);
#endif
}
__device__ inline unsigned short bf1(float f) {
    return (unsigned short)(pack2bf(f, f) & 0xFFFFu);
}

// ---------------------------------------------------------------------------
// Generic bf16-WMMA GEMM.
//   B_NT = true :  C = act(scale * A(MxK) * B(NxK)^T + bias)
//   B_NT = false:  C = act(scale * A(MxK) * B(KxN)   + bias)
//
// 256 threads = 8 waves. Block tile 128x128, wave tile 32x64 (2x4 WMMA 16x16),
// K stepped by 32 via v_wmma_f32_16x16x32_bf16.
// Ping-pong LDS: bf16 tiles [row][k], lane half h reads k=16h..16h+15 (K-pairs
// packed per dword, row stride 18 dwords). Fragment build is pure ds_load
// (merges to ds_load_2addr_b64) feeding v_wmma directly. fp32->bf16 happens
// once at staging (1 VALU/dword). Per K-slab: store next slab into the other
// LDS buffer *before* compute (latency hidden under WMMA), single barrier.
// Batched via blockIdx.z: z -> (bb=z/batchH, hh=z%batchH) pointer offsets.
// ---------------------------------------------------------------------------
#define LDS_STRIDE 18   // dwords per row (16 data dwords = 32 bf16, +2 pad)

template<bool B_NT>
__global__ void __launch_bounds__(256)
gemm_wmma(const float* __restrict__ A, const float* __restrict__ Bm,
          const float* __restrict__ bias, float* __restrict__ C,
          int M, int N, int K, int lda, int ldb, int ldc,
          float scale, int act, int batchH,
          long long sAb, long long sAh, long long sBb, long long sBh,
          long long sCb, long long sCh)
{
    int z  = blockIdx.z;
    int bb = z / batchH, hh = z - bb * batchH;
    A  += (long long)bb * sAb + (long long)hh * sAh;
    Bm += (long long)bb * sBb + (long long)hh * sBh;
    C  += (long long)bb * sCb + (long long)hh * sCh;

    __shared__ unsigned AsU[2][128 * LDS_STRIDE];
    __shared__ unsigned BsU[2][128 * LDS_STRIDE];

    const int m0  = blockIdx.y * 128;
    const int n0  = blockIdx.x * 128;
    const int tid = threadIdx.x;
    const int wave = tid >> 5;
    const int lane = tid & 31;
    const int wm = (wave >> 1) * 32;   // 0,32,64,96
    const int wn = (wave & 1) * 64;    // 0,64
    const int lrow = lane & 15;
    const int hi   = lane >> 4;        // K-half select

    float4 ra[4], rb[4];

    // ---- global -> registers; 'full' is uniform (tail only on last slab) ----
    auto load_a = [&](int k0, bool full) {
        #pragma unroll
        for (int it = 0; it < 4; ++it) {
            int i = tid + it * 256;            // 0..1023
            int row = i >> 3, c4 = (i & 7) * 4;
            int gm = m0 + row, gk = k0 + c4;
            float4 v = make_float4(0.f, 0.f, 0.f, 0.f);
            if (gm < M) {
                if (full || gk + 3 < K) {
                    v = *(const float4*)(A + (long long)gm * lda + gk);
                } else {
                    float t0 = (gk     < K) ? A[(long long)gm * lda + gk    ] : 0.f;
                    float t1 = (gk + 1 < K) ? A[(long long)gm * lda + gk + 1] : 0.f;
                    float t2 = (gk + 2 < K) ? A[(long long)gm * lda + gk + 2] : 0.f;
                    float t3 = (gk + 3 < K) ? A[(long long)gm * lda + gk + 3] : 0.f;
                    v = make_float4(t0, t1, t2, t3);
                }
            }
            ra[it] = v;
        }
    };
    auto load_b = [&](int k0, bool full) {
        if constexpr (B_NT) {
            #pragma unroll
            for (int it = 0; it < 4; ++it) {
                int i = tid + it * 256;
                int row = i >> 3, c4 = (i & 7) * 4;
                int gn = n0 + row, gk = k0 + c4;
                float4 v = make_float4(0.f, 0.f, 0.f, 0.f);
                if (gn < N) {
                    if (full || gk + 3 < K) {
                        v = *(const float4*)(Bm + (long long)gn * ldb + gk);
                    } else {
                        float t0 = (gk     < K) ? Bm[(long long)gn * ldb + gk    ] : 0.f;
                        float t1 = (gk + 1 < K) ? Bm[(long long)gn * ldb + gk + 1] : 0.f;
                        float t2 = (gk + 2 < K) ? Bm[(long long)gn * ldb + gk + 2] : 0.f;
                        float t3 = (gk + 3 < K) ? Bm[(long long)gn * ldb + gk + 3] : 0.f;
                        v = make_float4(t0, t1, t2, t3);
                    }
                }
                rb[it] = v;
            }
        } else {
            #pragma unroll
            for (int it = 0; it < 4; ++it) {
                int i = tid + it * 256;            // 32 k-rows x 32 n-quads
                int row = i >> 5, c4 = (i & 31) * 4;
                int gk = k0 + row, gn = n0 + c4;
                float4 v = make_float4(0.f, 0.f, 0.f, 0.f);
                if (full || gk < K) {
                    if (gn + 3 < N) {
                        v = *(const float4*)(Bm + (long long)gk * ldb + gn);
                    } else {
                        float t0 = (gn     < N) ? Bm[(long long)gk * ldb + gn    ] : 0.f;
                        float t1 = (gn + 1 < N) ? Bm[(long long)gk * ldb + gn + 1] : 0.f;
                        float t2 = (gn + 2 < N) ? Bm[(long long)gk * ldb + gn + 2] : 0.f;
                        float t3 = (gn + 3 < N) ? Bm[(long long)gk * ldb + gn + 3] : 0.f;
                        v = make_float4(t0, t1, t2, t3);
                    }
                }
                rb[it] = v;
            }
        }
    };
    // ---- registers -> LDS: 1 v_perm/v_cvt_pk per dword + ds_store_b64 ----
    auto store_ab = [&](int p) {
        #pragma unroll
        for (int it = 0; it < 4; ++it) {
            int i = tid + it * 256;
            int row = i >> 3, c4 = i & 7;
            int di = row * LDS_STRIDE + c4 * 2;      // always even -> 8B aligned
            *(uint2*)&AsU[p][di] = make_uint2(pack2bf(ra[it].x, ra[it].y),
                                              pack2bf(ra[it].z, ra[it].w));
        }
        if constexpr (B_NT) {
            #pragma unroll
            for (int it = 0; it < 4; ++it) {
                int i = tid + it * 256;
                int row = i >> 3, c4 = i & 7;
                int di = row * LDS_STRIDE + c4 * 2;
                *(uint2*)&BsU[p][di] = make_uint2(pack2bf(rb[it].x, rb[it].y),
                                                  pack2bf(rb[it].z, rb[it].w));
            }
        } else {
            unsigned short* BsH = (unsigned short*)BsU[p];
            #pragma unroll
            for (int it = 0; it < 4; ++it) {
                int i = tid + it * 256;
                int row = i >> 5, nl = (i & 31) * 4;   // row = local k
                BsH[(nl    ) * (2 * LDS_STRIDE) + row] = bf1(rb[it].x);
                BsH[(nl + 1) * (2 * LDS_STRIDE) + row] = bf1(rb[it].y);
                BsH[(nl + 2) * (2 * LDS_STRIDE) + row] = bf1(rb[it].z);
                BsH[(nl + 3) * (2 * LDS_STRIDE) + row] = bf1(rb[it].w);
            }
        }
    };

    v8f acc[2][4] = {};

    // ---- pipeline prologue: slab 0 -> LDS buf0; slab 1 -> registers ----
    {
        bool full0 = (32 <= K);
        load_a(0, full0); load_b(0, full0);
        store_ab(0);
    }
    __syncthreads();
    if (32 < K) {
        bool full1 = (64 <= K);
        load_a(32, full1); load_b(32, full1);
    }

    int p = 0;
    for (int k0 = 0;;) {
        int k1 = k0 + 32;
        bool more = (k1 < K);
        if (more) {
            store_ab(p ^ 1);                     // regs hold slab k1 -> other buffer
            if (k1 + 32 < K) {                   // begin loading slab k1+32
                bool full2 = (k1 + 64 <= K);
                load_a(k1 + 32, full2); load_b(k1 + 32, full2);
                __builtin_prefetch(A  + (long long)(m0 + (tid >> 1)) * lda + k1 + 64, 0, 1);
                __builtin_prefetch(Bm + (long long)(n0 + (tid >> 1)) * (B_NT ? ldb : 1)
                                      + (B_NT ? (k1 + 64) : (long long)(k1 + 64) * ldb), 0, 1);
            }
        }

        // ---- fragments straight out of LDS buf[p] (pure ds loads) ----
        BF16Frag bfr[4];
        #pragma unroll
        for (int jj = 0; jj < 4; ++jj) {
            const unsigned* bp = &BsU[p][(wn + jj * 16 + lrow) * LDS_STRIDE + hi * 8];
            #pragma unroll
            for (int vv = 0; vv < 8; ++vv) bfr[jj].u[vv] = bp[vv];
        }
        #pragma unroll
        for (int ii = 0; ii < 2; ++ii) {
            const unsigned* ap = &AsU[p][(wm + ii * 16 + lrow) * LDS_STRIDE + hi * 8];
            BF16Frag af;
            af.u[0] = ap[0]; af.u[1] = ap[1]; af.u[2] = ap[2];  af.u[3] = ap[3];
            af.u[4] = ap[4]; af.u[5] = ap[5]; af.u[6] = ap[6]; af.u[7] = ap[7];
            #pragma unroll
            for (int jj = 0; jj < 4; ++jj)
                acc[ii][jj] = __builtin_amdgcn_wmma_f32_16x16x32_bf16(
                    false, af.v, false, bfr[jj].v, (short)0, acc[ii][jj], false, false);
        }

        if (!more) break;
        __syncthreads();     // single barrier per slab (ping-pong)
        p ^= 1;
        k0 = k1;
    }

    // ---- epilogue: scale, bias, activation, guarded store ----
    #pragma unroll
    for (int ii = 0; ii < 2; ++ii)
        #pragma unroll
        for (int jj = 0; jj < 4; ++jj)
            #pragma unroll
            for (int r = 0; r < 8; ++r) {
                int row = m0 + wm + ii * 16 + hi * 8 + r;
                int col = n0 + wn + jj * 16 + lrow;
                if (row < M && col < N) {
                    float v = acc[ii][jj][r] * scale + (bias ? bias[col] : 0.f);
                    if (act == 1)      v = 0.5f * v * (1.f + erff(v * 0.70710678118654752f));
                    else if (act == 2) v = tanhf(v);
                    C[(long long)row * ldc + col] = v;
                }
            }
}

// ---------------------------------------------------------------------------
// Attention-1: per (b,h), head-dim 16, causal, online softmax. grid (H,B), 256 thr
// ---------------------------------------------------------------------------
__global__ void __launch_bounds__(256)
attn1_kernel(const float* __restrict__ q, const float* __restrict__ k,
             const float* __restrict__ v, float* __restrict__ att)
{
    int h = blockIdx.x, b = blockIdx.y;
    __shared__ float Ks[250 * 16];
    __shared__ float Vs[250 * 16];
    int tid = threadIdx.x;
    for (int i = tid; i < 250 * 16; i += 256) {
        int r = i >> 4, c = i & 15;
        long long g = ((long long)(b * 250 + r)) * 128 + h * 16 + c;
        Ks[i] = k[g];
        Vs[i] = v[g];
    }
    __syncthreads();
    int i = tid;
    if (i < 250) {
        float qr[16], acc[16];
        long long gq = ((long long)(b * 250 + i)) * 128 + h * 16;
        #pragma unroll
        for (int c = 0; c < 16; ++c) { qr[c] = q[gq + c]; acc[c] = 0.f; }
        float m = -3.0e38f, ssum = 0.f;
        for (int j = 0; j <= i; ++j) {
            float s = 0.f;
            #pragma unroll
            for (int c = 0; c < 16; ++c) s += qr[c] * Ks[j * 16 + c];
            s *= 0.25f;                       // 1/sqrt(16)
            float nm = fmaxf(m, s);
            float f = __expf(m - nm), p = __expf(s - nm);
            ssum = ssum * f + p;
            #pragma unroll
            for (int c = 0; c < 16; ++c) acc[c] = acc[c] * f + p * Vs[j * 16 + c];
            m = nm;
        }
        float inv = 1.f / ssum;
        #pragma unroll
        for (int c = 0; c < 16; ++c) att[gq + c] = acc[c] * inv;
    }
}

// ---------------------------------------------------------------------------
// out = LayerNorm(a + res) over D=128. grid rows, block 128.
// ---------------------------------------------------------------------------
__global__ void __launch_bounds__(128)
add_ln_kernel(const float* __restrict__ a, const float* __restrict__ res,
              const float* __restrict__ g, const float* __restrict__ be,
              float* __restrict__ out)
{
    int row = blockIdx.x, t = threadIdx.x;
    __shared__ float red[128];
    float v = a[(long long)row * 128 + t] + res[(long long)row * 128 + t];
    red[t] = v; __syncthreads();
    for (int s = 64; s > 0; s >>= 1) { if (t < s) red[t] += red[t + s]; __syncthreads(); }
    float mean = red[0] * (1.f / 128.f); __syncthreads();
    float dv = v - mean;
    red[t] = dv * dv; __syncthreads();
    for (int s = 64; s > 0; s >>= 1) { if (t < s) red[t] += red[t + s]; __syncthreads(); }
    float var = red[0] * (1.f / 128.f);
    out[(long long)row * 128 + t] = dv * rsqrtf(var + 1e-5f) * g[t] + be[t];
}

// gc[b][d] = mean_s h[b][s][d]. grid B, block 128.
__global__ void mean_kernel(const float* __restrict__ h, float* __restrict__ gc)
{
    int b = blockIdx.x, d = threadIdx.x;
    float s = 0.f;
    for (int i = 0; i < 250; ++i) s += h[((long long)b * 250 + i) * 128 + d];
    gc[b * 128 + d] = s * (1.f / 250.f);
}

// Row softmax on padded (rows x 256) score buffer, 250 valid cols.
__global__ void __launch_bounds__(256)
softmax_kernel(float* __restrict__ sc)
{
    long long row = blockIdx.x;
    float* p = sc + row * 256;
    int t = threadIdx.x;
    __shared__ float red[256];
    float v = (t < 250) ? p[t] : -3.0e38f;
    red[t] = v; __syncthreads();
    for (int s = 128; s > 0; s >>= 1) { if (t < s) red[t] = fmaxf(red[t], red[t + s]); __syncthreads(); }
    float mx = red[0]; __syncthreads();
    float e = (t < 250) ? __expf(v - mx) : 0.f;
    red[t] = e; __syncthreads();
    for (int s = 128; s > 0; s >>= 1) { if (t < s) red[t] += red[t + s]; __syncthreads(); }
    float inv = 1.f / red[0];
    p[t] = e * inv;   // pad cols get 0 (never read: K-guard in gemm)
}

// ---------------------------------------------------------------------------
// Depthwise dilated conv + adaptive gating. grid (ND=8, B), block 128 (=d).
// cat[b][s][ci*128+d] = (sum_t w[t]*h[b][s+t*dil-pad][d] + bias) * aw[b][ci*128+d]
// ---------------------------------------------------------------------------
template<int KW>
__global__ void __launch_bounds__(128)
dwconv_gate(const float* __restrict__ h, const float* __restrict__ W,
            const float* __restrict__ bconv, const float* __restrict__ aw,
            float* __restrict__ cat, int kiBase)
{
    int di = blockIdx.x, b = blockIdx.y, d = threadIdx.x;
    int dil = di + 1;
    int pad = ((KW - 1) / 2) * dil;
    int ci = kiBase + di;
    float w[KW];
    #pragma unroll
    for (int t = 0; t < KW; ++t) w[t] = W[(di * 128 + d) * KW + t];
    float bs = bconv[di * 128 + d];
    float gt = aw[(long long)b * 5120 + ci * 128 + d];
    const float* hb = h + (long long)b * 250 * 128 + d;
    float* cb = cat + (long long)b * 250 * 5120 + ci * 128 + d;
    for (int s = 0; s < 250; ++s) {
        float y = 0.f;
        #pragma unroll
        for (int t = 0; t < KW; ++t) {
            int si = s + t * dil - pad;
            if (si >= 0 && si < 250) y += w[t] * hb[(long long)si * 128];
        }
        cb[(long long)s * 5120] = (y + bs) * gt;
    }
}

// ---------------------------------------------------------------------------
// Host orchestration
// ---------------------------------------------------------------------------
extern "C" void kernel_launch(void* const* d_in, const int* in_sizes, int n_in,
                              void* d_out, int out_size, void* d_ws, size_t ws_size,
                              hipStream_t stream) {
    (void)in_sizes; (void)n_in; (void)out_size; (void)ws_size;
    const float* x    = (const float*)d_in[0];
    const float* Wq   = (const float*)d_in[1];
    const float* Wk   = (const float*)d_in[2];
    const float* Wv   = (const float*)d_in[3];
    const float* Wo   = (const float*)d_in[4];
    const float* bq   = (const float*)d_in[5];
    const float* bk   = (const float*)d_in[6];
    const float* bv   = (const float*)d_in[7];
    const float* bo   = (const float*)d_in[8];
    const float* ln1g = (const float*)d_in[9];
    const float* ln1b = (const float*)d_in[10];
    const float* Wg1  = (const float*)d_in[11];
    const float* bg1  = (const float*)d_in[12];
    const float* Wg2  = (const float*)d_in[13];
    const float* bg2  = (const float*)d_in[14];
    const float* dwW[5] = { (const float*)d_in[15], (const float*)d_in[17],
                            (const float*)d_in[19], (const float*)d_in[21],
                            (const float*)d_in[23] };
    const float* dwb[5] = { (const float*)d_in[16], (const float*)d_in[18],
                            (const float*)d_in[20], (const float*)d_in[22],
                            (const float*)d_in[24] };
    const float* Wi   = (const float*)d_in[25];
    const float* bi   = (const float*)d_in[26];
    const float* Wow  = (const float*)d_in[27];
    const float* bow  = (const float*)d_in[28];
    const float* Wop  = (const float*)d_in[29];
    const float* bop  = (const float*)d_in[30];
    const float* ln2g = (const float*)d_in[31];
    const float* ln2b = (const float*)d_in[32];

    float* ws = (float*)d_ws;
    const long long SZX = 16LL * 250 * 128;   // 512000
    float* Q    = ws;
    float* Kb   = Q + SZX;
    float* V    = Kb + SZX;
    float* ATT  = V + SZX;
    float* TMP1 = ATT + SZX;
    float* H    = TMP1 + SZX;
    float* GC   = H + SZX;
    float* G1   = GC + 2048;
    float* AW   = G1 + 2048;
    float* FIN  = AW + 16LL * 5120;
    float* CAT  = FIN + SZX;                    // aliased later as ATT2
    float* SC   = CAT + 16LL * 250 * 5120;
    float* QKV2 = SC + 128LL * 250 * 256;       // aliased later as PROJ2
    float* ATT2  = CAT;
    float* PROJ2 = QKV2;

    const dim3 blk(256);
    const int MT = 32;                          // ceil(4000/128)

    // 1) q,k,v = x @ W{q,k,v}^T + b   (M=4000, N=128, K=128)
    gemm_wmma<true><<<dim3(1, MT, 1), blk, 0, stream>>>(x, Wq, bq, Q, 4000, 128, 128,
        128, 128, 128, 1.f, 0, 1, 0, 0, 0, 0, 0, 0);
    gemm_wmma<true><<<dim3(1, MT, 1), blk, 0, stream>>>(x, Wk, bk, Kb, 4000, 128, 128,
        128, 128, 128, 1.f, 0, 1, 0, 0, 0, 0, 0, 0);
    gemm_wmma<true><<<dim3(1, MT, 1), blk, 0, stream>>>(x, Wv, bv, V, 4000, 128, 128,
        128, 128, 128, 1.f, 0, 1, 0, 0, 0, 0, 0, 0);

    // 2) causal attention (hd=16)
    attn1_kernel<<<dim3(8, 16, 1), dim3(256), 0, stream>>>(Q, Kb, V, ATT);

    // 3) tmp1 = att @ Wo^T + bo ; 4) h = LN(tmp1 + x)
    gemm_wmma<true><<<dim3(1, MT, 1), blk, 0, stream>>>(ATT, Wo, bo, TMP1, 4000, 128, 128,
        128, 128, 128, 1.f, 0, 1, 0, 0, 0, 0, 0, 0);
    add_ln_kernel<<<dim3(4000), dim3(128), 0, stream>>>(TMP1, x, ln1g, ln1b, H);

    // 5) gc = mean_s h ; 6) g1 = gelu(gc@Wg1^T+bg1) ; 7) aw = tanh(g1@Wg2^T+bg2)
    mean_kernel<<<dim3(16), dim3(128), 0, stream>>>(H, GC);
    gemm_wmma<true><<<dim3(1, 1, 1), blk, 0, stream>>>(GC, Wg1, bg1, G1, 16, 128, 128,
        128, 128, 128, 1.f, 1, 1, 0, 0, 0, 0, 0, 0);
    gemm_wmma<true><<<dim3(40, 1, 1), blk, 0, stream>>>(G1, Wg2, bg2, AW, 16, 5120, 128,
        128, 128, 5120, 1.f, 2, 1, 0, 0, 0, 0, 0, 0);

    // 8) 40 gated dilated depthwise convs -> cat (B,S,5120)
    dwconv_gate<3><<<dim3(8, 16, 1), dim3(128), 0, stream>>>(H, dwW[0], dwb[0], AW, CAT, 0);
    dwconv_gate<5><<<dim3(8, 16, 1), dim3(128), 0, stream>>>(H, dwW[1], dwb[1], AW, CAT, 8);
    dwconv_gate<7><<<dim3(8, 16, 1), dim3(128), 0, stream>>>(H, dwW[2], dwb[2], AW, CAT, 16);
    dwconv_gate<9><<<dim3(8, 16, 1), dim3(128), 0, stream>>>(H, dwW[3], dwb[3], AW, CAT, 24);
    dwconv_gate<11><<<dim3(8, 16, 1), dim3(128), 0, stream>>>(H, dwW[4], dwb[4], AW, CAT, 32);

    // 9) qkv2 = cat @ Wi^T + bi  (M=4000, N=15360, K=5120) -- dominant GEMM
    gemm_wmma<true><<<dim3(120, MT, 1), blk, 0, stream>>>(CAT, Wi, bi, QKV2,
        4000, 15360, 5120, 5120, 5120, 15360, 1.f, 0, 1, 0, 0, 0, 0, 0, 0);

    // 10) scores = Q2 @ K2^T / sqrt(640), batched over (b,h); ld padded to 256
    gemm_wmma<true><<<dim3(2, 2, 128), blk, 0, stream>>>(QKV2, QKV2 + 5120, nullptr, SC,
        250, 250, 640, 15360, 15360, 256, 0.039528470752104741f, 0, 8,
        250LL * 15360, 640, 250LL * 15360, 640, 8LL * 250 * 256, 250LL * 256);

    // 11) row softmax (no mask for MHA2)
    softmax_kernel<<<dim3(32000), dim3(256), 0, stream>>>(SC);

    // 12) att2 = P @ V2 (NN GEMM), batched; writes (B,S,5120) into CAT region
    gemm_wmma<false><<<dim3(5, 2, 128), blk, 0, stream>>>(SC, QKV2 + 10240, nullptr, ATT2,
        250, 640, 250, 256, 15360, 5120, 1.f, 0, 8,
        8LL * 250 * 256, 250LL * 256, 250LL * 15360, 640, 250LL * 5120, 640);

    // 13) proj2 = att2 @ Wow^T + bow  (reuses QKV2 space)
    gemm_wmma<true><<<dim3(40, MT, 1), blk, 0, stream>>>(ATT2, Wow, bow, PROJ2,
        4000, 5120, 5120, 5120, 5120, 5120, 1.f, 0, 1, 0, 0, 0, 0, 0, 0);

    // 14) fin = proj2 @ Wop^T + bop ; 15) out = LN(fin + h)
    gemm_wmma<true><<<dim3(1, MT, 1), blk, 0, stream>>>(PROJ2, Wop, bop, FIN,
        4000, 128, 5120, 5120, 5120, 128, 1.f, 0, 1, 0, 0, 0, 0, 0, 0);
    add_ln_kernel<<<dim3(4000), dim3(128), 0, stream>>>(FIN, H, ln2g, ln2b, (float*)d_out);
}
